// MultiheadSelfAttention_29480655519994
// MI455X (gfx1250) — matrix operand order repaired and checked by reading, not compiled, not run on hardware
//
#include <hip/hip_runtime.h>

typedef __attribute__((ext_vector_type(16))) __bf16 bf16x16;
typedef __attribute__((ext_vector_type(8)))  float  f32x8;

#define WS   7
#define WSQ  49
#define NH   8
#define HD   32
#define CIN  256
#define NQKV 768
#define SCALE_F 5.656854249492380f   // sqrt(32): reference multiplies scores by sqrt(hd)

__device__ __forceinline__ unsigned short f2bf(float f) {
  unsigned u = __float_as_uint(f);
  u += 0x7FFFu + ((u >> 16) & 1u);     // round-to-nearest-even
  return (unsigned short)(u >> 16);
}

__device__ __forceinline__ f32x8 fzero() {
  f32x8 z = {0.f, 0.f, 0.f, 0.f, 0.f, 0.f, 0.f, 0.f};
  return z;
}

union FragU { unsigned u[8]; bf16x16 v; };

// A-matrix 16x32 bf16 fragment per ISA layout:
// lanes 0-15: M=lane, VGPR0..3 = K 0..7, VGPR4..7 = K 16..23
// lanes 16-31: M=lane-16, K base +8 (8..15 / 24..31)
__device__ __forceinline__ bf16x16 load_a_frag(const unsigned short* p, int S, int mr, int kc) {
  int lane = threadIdx.x & 31;
  const unsigned short* row = p + (mr + (lane & 15)) * S + kc + ((lane >> 4) << 3);
  FragU f;
#pragma unroll
  for (int j = 0; j < 4; ++j) {
    f.u[j]     = *(const unsigned*)(row + 2 * j);
    f.u[j + 4] = *(const unsigned*)(row + 16 + 2 * j);
  }
  return f.v;
}

// B-matrix 32x16 bf16 fragment, read from N-major (transposed) LDS tile Bt[n][k]:
// lanes 0-15: N=lane, VGPR0..3 = K 0..7, VGPR4..7 = K 16..23; lanes 16-31: N=lane-16, K base +8 (8..15 / 24..31)
__device__ __forceinline__ bf16x16 load_b_frag(const unsigned short* p, int S, int nb, int kc) {
  int lane = threadIdx.x & 31;
  const unsigned short* row = p + (nb + (lane & 15)) * S + kc + ((lane >> 4) << 3);
  FragU f;
#pragma unroll
  for (int j = 0; j < 8; ++j) f.u[j] = *(const unsigned*)(row + ((j >> 2) << 4) + 2 * (j & 3));
  return f.v;
}

__device__ __forceinline__ f32x8 wmma_bf16(bf16x16 a, bf16x16 b, f32x8 c) {
  return __builtin_amdgcn_wmma_f32_16x16x32_bf16(false, a, false, b, (short)0, c, false, false);
}

// ---------------------------------------------------------------------------
// Kernel 1: fused QKV GEMM + windowed attention. One block per 7x7 window.
// 256 threads = 8 wave32; phase 2 assigns one head per wave.
// ---------------------------------------------------------------------------
__global__ __launch_bounds__(256) void win_attn_kernel(
    const float* __restrict__ x, const float* __restrict__ pos,
    const float* __restrict__ wqkv, const float* __restrict__ bqkv,
    unsigned short* __restrict__ attn_out) {
  __shared__ __align__(16) unsigned short Xs[64 * 256];     // 32 KB, x tile (also attn scratch in phase 2)
  __shared__ __align__(16) unsigned short Bt[32 * 256];     // 16 KB, w_qkv column chunk (transposed)
  __shared__ __align__(16) unsigned short Qs[NH * 64 * HD]; // 32 KB
  __shared__ __align__(16) unsigned short Ks[NH * 64 * HD]; // 32 KB
  __shared__ __align__(16) unsigned short Vt[NH * HD * 64]; // 32 KB  (V transposed: [d][l])

  const int tid  = threadIdx.x;
  const int lane = tid & 31;
  const int wv   = tid >> 5;
  const int wi   = blockIdx.x;
  const int b    = wi >> 6;
  const int wh   = (wi >> 3) & 7;
  const int ww   = wi & 7;

  // ---- phase 0: stage window tokens (49x256) as bf16, zero-pad rows to 64 ----
  for (int idx = tid; idx < 64 * 256; idx += 256) {
    int m = idx >> 8, k = idx & 255;
    float v = 0.f;
    if (m < WSQ) {
      int h = wh * WS + m / WS;
      int w = ww * WS + m % WS;
      v = x[(size_t)((b * 56 + h) * 56 + w) * CIN + k];
    }
    Xs[idx] = f2bf(v);
  }

  // ---- phase 1: QKV = Xs @ Wqkv + b, 24 column-chunks of 32; write to LDS ----
  const int mr  = (wv & 3) << 4;  // wave M tile: 0/16/32/48
  const int nbl = (wv >> 2) << 4; // wave N tile within chunk: 0/16
  for (int c = 0; c < NQKV / 32; ++c) {
    __syncthreads();
    for (int idx = tid; idx < 32 * 256; idx += 256) {
      int n = idx & 31, k = idx >> 5;                 // consecutive lanes -> consecutive n (coalesced)
      Bt[n * 256 + k] = f2bf(wqkv[k * NQKV + c * 32 + n]);
    }
    __syncthreads();
    f32x8 acc = fzero();
#pragma unroll
    for (int kc = 0; kc < CIN / 32; ++kc) {
      bf16x16 a  = load_a_frag(Xs, 256, mr, kc * 32);
      bf16x16 bf = load_b_frag(Bt, 256, nbl, kc * 32);
      acc = wmma_bf16(a, bf, acc);
    }
    int n_g  = c * 32 + nbl + (lane & 15);
    int s    = n_g >> 8;         // 0=q 1=k 2=v
    int head = (n_g >> 5) & 7;
    int d    = n_g & 31;
    float bq = bqkv[n_g];
#pragma unroll
    for (int r = 0; r < 8; ++r) {
      int m = mr + r + ((lane >> 4) << 3);            // C-fragment row
      unsigned short bv = f2bf(acc[r] + bq);
      if (s == 0)      Qs[(head * 64 + m) * HD + d] = bv;
      else if (s == 1) Ks[(head * 64 + m) * HD + d] = bv;
      else             Vt[(head * HD + d) * 64 + m] = bv;
    }
  }
  __syncthreads();

  // ---- phase 2: attention; wave wv handles head wv ----
  const unsigned short* Qh = Qs + wv * 64 * HD;
  const unsigned short* Kh = Ks + wv * 64 * HD;     // [l][d]: already "transposed-for-B"
  const unsigned short* Vh = Vt + wv * HD * 64;     // [d][l]
  unsigned short* attw = Xs + wv * 16 * 64;         // per-wave scratch (Xs is dead)

  f32x8 sc[4][4];
  bf16x16 kb[4];
#pragma unroll
  for (int nt = 0; nt < 4; ++nt) kb[nt] = load_b_frag(Kh, HD, nt * 16, 0);
#pragma unroll
  for (int mt = 0; mt < 4; ++mt) {
    bf16x16 a = load_a_frag(Qh, HD, mt * 16, 0);
#pragma unroll
    for (int nt = 0; nt < 4; ++nt) sc[mt][nt] = wmma_bf16(a, kb[nt], fzero());
  }

  const int ncol = lane & 15;
  const int moff = (lane >> 4) << 3;

  // scale + relative position bias + mask padding
#pragma unroll
  for (int mt = 0; mt < 4; ++mt) {
#pragma unroll
    for (int nt = 0; nt < 4; ++nt) {
      int n = nt * 16 + ncol;
#pragma unroll
      for (int r = 0; r < 8; ++r) {
        int m = mt * 16 + r + moff;
        float v = sc[mt][nt][r];
        if (m < WSQ && n < WSQ)
          v = v * SCALE_F + pos[(m / WS + m % WS) * 13 + (n / WS + n % WS)];
        else
          v = -3.0e38f;
        sc[mt][nt][r] = v;
      }
    }
  }

  // row softmax: reduce over 4 N-tiles locally + 16 lanes via shfl_xor (wave32)
#pragma unroll
  for (int mt = 0; mt < 4; ++mt) {
#pragma unroll
    for (int r = 0; r < 8; ++r) {
      float mx = sc[mt][0][r];
#pragma unroll
      for (int nt = 1; nt < 4; ++nt) mx = fmaxf(mx, sc[mt][nt][r]);
#pragma unroll
      for (int off = 1; off < 16; off <<= 1) mx = fmaxf(mx, __shfl_xor(mx, off, 32));
      float sum = 0.f;
#pragma unroll
      for (int nt = 0; nt < 4; ++nt) {
        float e = __expf(sc[mt][nt][r] - mx);
        sc[mt][nt][r] = e;
        sum += e;
      }
#pragma unroll
      for (int off = 1; off < 16; off <<= 1) sum += __shfl_xor(sum, off, 32);
      float inv = 1.f / sum;
#pragma unroll
      for (int nt = 0; nt < 4; ++nt) sc[mt][nt][r] *= inv;
    }
  }

  // attn @ V, one 16-row strip at a time (C-layout -> A-layout via LDS scratch)
#pragma unroll
  for (int mt = 0; mt < 4; ++mt) {
#pragma unroll
    for (int nt = 0; nt < 4; ++nt) {
#pragma unroll
      for (int r = 0; r < 8; ++r)
        attw[(r + moff) * 64 + nt * 16 + ncol] = f2bf(sc[mt][nt][r]);
    }
    __syncthreads();
    f32x8 o0 = fzero(), o1 = fzero();
#pragma unroll
    for (int kc = 0; kc < 2; ++kc) {
      bf16x16 a = load_a_frag(attw, 64, 0, kc * 32);
      o0 = wmma_bf16(a, load_b_frag(Vh, 64, 0,  kc * 32), o0);
      o1 = wmma_bf16(a, load_b_frag(Vh, 64, 16, kc * 32), o1);
    }
    __syncthreads();
#pragma unroll
    for (int r = 0; r < 8; ++r) {
      int m = mt * 16 + r + moff;
      if (m < WSQ) {
        int h = wh * WS + m / WS;
        int w = ww * WS + m % WS;
        size_t tok = (size_t)(b * 56 + h) * 56 + w;
        unsigned short* dst = attn_out + tok * (NH * HD) + wv * HD;
        dst[ncol]      = f2bf(o0[r]);
        dst[ncol + 16] = f2bf(o1[r]);
      }
    }
  }
}

// ---------------------------------------------------------------------------
// Kernel 2: out = attn_out(bf16) @ w_out + b_out  (M x 256 @ 256 x 256, f32 out)
// Block tile 64x128, 8 waves each 32x32, K chunks of 32.
// ---------------------------------------------------------------------------
__global__ __launch_bounds__(256) void out_proj_kernel(
    const unsigned short* __restrict__ ain, const float* __restrict__ wout,
    const float* __restrict__ bout, float* __restrict__ out) {
  __shared__ __align__(16) unsigned short As[64 * 34];
  __shared__ __align__(16) unsigned short Bt[128 * 34];
  const int tid  = threadIdx.x;
  const int lane = tid & 31;
  const int wv   = tid >> 5;
  const int bm   = blockIdx.x >> 1;
  const int bn   = blockIdx.x & 1;
  const int mr   = (wv & 1) << 5;
  const int nb   = (wv >> 1) << 5;

  f32x8 acc[2][2];
#pragma unroll
  for (int i = 0; i < 2; ++i)
#pragma unroll
    for (int j = 0; j < 2; ++j) acc[i][j] = fzero();

  for (int kc = 0; kc < 8; ++kc) {
    __syncthreads();
    for (int idx = tid; idx < 64 * 32; idx += 256) {
      int m = idx >> 5, k = idx & 31;
      As[m * 34 + k] = ain[(size_t)(bm * 64 + m) * 256 + kc * 32 + k];
    }
    for (int idx = tid; idx < 128 * 32; idx += 256) {
      int n = idx & 127, k = idx >> 7;                // consecutive lanes -> consecutive n
      Bt[n * 34 + k] = f2bf(wout[(kc * 32 + k) * 256 + bn * 128 + n]);
    }
    __syncthreads();
#pragma unroll
    for (int i = 0; i < 2; ++i) {
      bf16x16 a = load_a_frag(As, 34, mr + 16 * i, 0);
#pragma unroll
      for (int j = 0; j < 2; ++j) {
        bf16x16 bf = load_b_frag(Bt, 34, nb + 16 * j, 0);
        acc[i][j] = wmma_bf16(a, bf, acc[i][j]);
      }
    }
  }
  const int ncol = lane & 15;
  const int moff = (lane >> 4) << 3;
#pragma unroll
  for (int i = 0; i < 2; ++i) {
#pragma unroll
    for (int j = 0; j < 2; ++j) {
      int n = bn * 128 + nb + 16 * j + ncol;
      float bo = bout[n];
#pragma unroll
      for (int r = 0; r < 8; ++r) {
        int m = bm * 64 + mr + 16 * i + r + moff;
        out[(size_t)m * 256 + n] = acc[i][j][r] + bo;
      }
    }
  }
}

extern "C" void kernel_launch(void* const* d_in, const int* in_sizes, int n_in,
                              void* d_out, int out_size, void* d_ws, size_t ws_size,
                              hipStream_t stream) {
  const float* x     = (const float*)d_in[0];
  const float* pos   = (const float*)d_in[1];
  const float* wqkv  = (const float*)d_in[2];
  const float* bqkv  = (const float*)d_in[3];
  const float* wout  = (const float*)d_in[4];
  const float* bout  = (const float*)d_in[5];
  float* out = (float*)d_out;

  const int B = in_sizes[0] / (56 * 56 * CIN);
  const int windows = B * 8 * 8;            // (56/7)^2 windows per image
  const int Mtok = B * 56 * 56;

  unsigned short* attn_ws = (unsigned short*)d_ws;  // Mtok * 256 bf16

  hipLaunchKernelGGL(win_attn_kernel, dim3(windows), dim3(256), 0, stream,
                     x, pos, wqkv, bqkv, attn_ws);
  hipLaunchKernelGGL(out_proj_kernel, dim3((Mtok / 64) * 2), dim3(256), 0, stream,
                     attn_ws, wout, bout, out);
}
